// Matcher_14353780703873
// MI455X (gfx1250) — matrix operand order repaired and checked by reading, not compiled, not run on hardware
//
#include <hip/hip_runtime.h>
#include <cstdint>

typedef __attribute__((ext_vector_type(16))) _Float16 v16h;
typedef __attribute__((ext_vector_type(8)))  _Float16 v8h;
typedef __attribute__((ext_vector_type(8)))  float    v8f;

// ---------------------------------------------------------------------------
// CDNA5 async copy: global -> LDS, 16 bytes per lane, tracked by ASYNCcnt.
// ---------------------------------------------------------------------------
__device__ __forceinline__ void async_copy_b128(uint32_t ldsByteAddr, const void* gptr) {
  asm volatile("global_load_async_to_lds_b128 %0, %1, off"
               :: "v"(ldsByteAddr), "v"((uint64_t)(uintptr_t)gptr)
               : "memory");
}
__device__ __forceinline__ void wait_async0() {
  asm volatile("s_wait_asynccnt 0x0" ::: "memory");
}

// ---------------------------------------------------------------------------
// tnorm[b][n] = sum_c tar[b][c][n]^2   (f32 source, coalesced across n)
// ---------------------------------------------------------------------------
__global__ void tnorm_kernel(const float* __restrict__ t, float* __restrict__ tn,
                             int B, int C, int N) {
  int gid = blockIdx.x * blockDim.x + threadIdx.x;
  if (gid >= B * N) return;
  int b = gid / N, n = gid % N;
  const float* p = t + (size_t)b * C * N + n;
  float s = 0.f;
  for (int c = 0; c < C; ++c) {
    float v = p[(size_t)c * N];
    s += v * v;
  }
  tn[gid] = s;
}

// ---------------------------------------------------------------------------
// One-shot transpose + f32->f16 convert:  in [B][C][N] f32 -> out [B][N][C] f16
// ---------------------------------------------------------------------------
__global__ __launch_bounds__(256) void transpose_f16_kernel(
    const float* __restrict__ in, _Float16* __restrict__ outT, int C, int N) {
  __shared__ _Float16 tile[32][33];
  const int b     = blockIdx.z;
  const int nBase = blockIdx.x * 32;
  const int cBase = blockIdx.y * 32;
  const int lane  = threadIdx.x & 31;
  const int row   = threadIdx.x >> 5;   // 0..7
  const float* src = in + (size_t)b * C * N;
#pragma unroll
  for (int j = 0; j < 4; ++j) {
    int cc = row + j * 8;
    tile[cc][lane] = (_Float16)src[(size_t)(cBase + cc) * N + nBase + lane];
  }
  __syncthreads();
  _Float16* dst = outT + (size_t)b * N * C;
#pragma unroll
  for (int j = 0; j < 4; ++j) {
    int nn = row + j * 8;
    dst[(size_t)(nBase + nn) * C + cBase + lane] = tile[lane][nn];
  }
}

// ---------------------------------------------------------------------------
// Fused distance-argmin via WMMA on pre-transposed f16 data.
//   ST,TT: [B][N][C] f16;  idxOut[b][m] = argmin_n(tnorm[n] - 2*dot(s_m,t_n))
// Block = 256 threads (8 waves): 128 rows (16/wave), sweeps all N columns in
// chunks of 64 (4 x 16-col WMMA tiles/wave). K panels are double-buffered in
// LDS and filled with GLOBAL_LOAD_ASYNC_TO_LDS_B128 (ASYNCcnt) so the async
// engine fills panel k+1 while WMMAs consume panel k.
// ---------------------------------------------------------------------------
__global__ __launch_bounds__(256) void nn_argmin_kernel(
    const _Float16* __restrict__ ST, const _Float16* __restrict__ TT,
    const float* __restrict__ tnorm, int* __restrict__ idxOut,
    int C, int N) {
  constexpr int ROWS = 128, COLS = 64, KT = 32;
  constexpr int LDH = KT + 8;                   // 40 halves = 80B pitch (16B aligned)
  __shared__ _Float16 ldsA[2][ROWS * LDH];      // [buf][row][k]
  __shared__ _Float16 ldsB[2][COLS * LDH];      // [buf][col][k]

  const int b        = blockIdx.y;
  const int rowBase  = blockIdx.x * ROWS;
  const int tid      = threadIdx.x;
  const int lane     = tid & 31;
  const int wave     = tid >> 5;
  const int laneN    = lane & 15;
  const int laneHalf = lane >> 4;

  const _Float16* Sb = ST + (size_t)b * N * C;
  const _Float16* Tb = TT + (size_t)b * N * C;
  const float*    TN = tnorm + (size_t)b * N;

  float bestVal[8];
  int   bestIdx[8];
#pragma unroll
  for (int r = 0; r < 8; ++r) { bestVal[r] = 3.0e38f; bestIdx[r] = 0; }

  const v8f vzero = {0.f, 0.f, 0.f, 0.f, 0.f, 0.f, 0.f, 0.f};

  // per-thread staging coordinates (8-half = 16B chunks)
  const int aRr0 = tid >> 2, aKc = (tid & 3) * 8;   // A: 2 chunks/thread
  const int bCc  = tid >> 2, bKc = (tid & 3) * 8;   // B: 1 chunk/thread

  // LDS byte addresses (flat LDS addr truncates to DS offset, ISA 10.2)
  uint32_t aLds[2], bLds[2];
  aLds[0] = (uint32_t)(uintptr_t)&ldsA[0][0];
  aLds[1] = (uint32_t)(uintptr_t)&ldsA[1][0];
  bLds[0] = (uint32_t)(uintptr_t)&ldsB[0][0];
  bLds[1] = (uint32_t)(uintptr_t)&ldsB[1][0];
  const uint32_t aOff0 = (uint32_t)((aRr0 * LDH + aKc) * 2);
  const uint32_t aOff1 = (uint32_t)(((aRr0 + 64) * LDH + aKc) * 2);
  const uint32_t bOff  = (uint32_t)((bCc * LDH + bKc) * 2);

  // A fragment addressing (16x32 f16 layout): half selects K base 0/8;
  // elems 0..7 -> K=base..base+7, 8..15 -> K=base+16..base+23
  const int aRow = wave * 16 + laneN;
  const int aK   = laneHalf ? 8 : 0;
  const int bK   = laneHalf ? 8 : 0;    // B: same K layout as A (base 0/8, +16 for elems 8..15)

  for (int nBase = 0; nBase < N; nBase += COLS) {
    v8f acc[4];
#pragma unroll
    for (int ct = 0; ct < 4; ++ct) acc[ct] = vzero;

    auto stage = [&](int buf, int kB) {
      async_copy_b128(aLds[buf] + aOff0, &Sb[(size_t)(rowBase + aRr0) * C + kB + aKc]);
      async_copy_b128(aLds[buf] + aOff1, &Sb[(size_t)(rowBase + aRr0 + 64) * C + kB + aKc]);
      async_copy_b128(bLds[buf] + bOff,  &Tb[(size_t)(nBase + bCc) * C + kB + bKc]);
    };

    // prologue: fill buffer 0 with the first K panel
    int cur = 0;
    stage(0, 0);
    wait_async0();
    __syncthreads();

    for (int kBase = 0; kBase < C; kBase += KT) {
      const bool hasNext = (kBase + KT) < C;
      if (hasNext) stage(cur ^ 1, kBase + KT);   // async fill of next panel

      const _Float16* lA = &ldsA[cur][0];
      const _Float16* lB = &ldsB[cur][0];

      v8h alo = *(const v8h*)&lA[aRow * LDH + aK];
      v8h ahi = *(const v8h*)&lA[aRow * LDH + aK + 16];
      v16h a;
#pragma unroll
      for (int i = 0; i < 8; ++i) { a[i] = alo[i]; a[i + 8] = ahi[i]; }

      // preload all 4 B fragments, then issue the 4 WMMAs back-to-back
      v16h bfr[4];
#pragma unroll
      for (int ct = 0; ct < 4; ++ct) {
        const int bCol = ct * 16 + laneN;
        v8h blo = *(const v8h*)&lB[bCol * LDH + bK];
        v8h bhi = *(const v8h*)&lB[bCol * LDH + bK + 16];
#pragma unroll
        for (int i = 0; i < 8; ++i) { bfr[ct][i] = blo[i]; bfr[ct][i + 8] = bhi[i]; }
      }
#pragma unroll
      for (int ct = 0; ct < 4; ++ct) {
        acc[ct] = __builtin_amdgcn_wmma_f32_16x16x32_f16(
            false, a, false, bfr[ct], (short)0, acc[ct], false, false);
      }

      wait_async0();     // my async writes to buf[cur^1] landed
      __syncthreads();   // everyone done reading buf[cur] / writing buf[cur^1]
      cur ^= 1;
    }

    // ---- fold slab into running per-row argmin.
    // C/D layout: VGPR r, lanes 0-15 -> row r, lanes 16-31 -> row r+8; col=laneN.
#pragma unroll
    for (int ct = 0; ct < 4; ++ct) {
      const int n = nBase + ct * 16 + laneN;
      const float tn = TN[n];
#pragma unroll
      for (int r = 0; r < 8; ++r) {
        float score = tn - 2.0f * acc[ct][r];
        if (score < bestVal[r]) { bestVal[r] = score; bestIdx[r] = n; }
      }
    }
  }

  // ---- argmin reduce across the 16 lanes of each half (tie -> smaller index)
#pragma unroll
  for (int r = 0; r < 8; ++r) {
    float v = bestVal[r];
    int   ix = bestIdx[r];
#pragma unroll
    for (int m = 8; m >= 1; m >>= 1) {
      float ov = __shfl_xor(v, m, 32);
      int   oi = __shfl_xor(ix, m, 32);
      if (ov < v || (ov == v && oi < ix)) { v = ov; ix = oi; }
    }
    if (laneN == 0) {
      int row = rowBase + wave * 16 + laneHalf * 8 + r;
      idxOut[(size_t)b * N + row] = ix;
    }
  }
}

// ---------------------------------------------------------------------------
// Output channels [0, 2048): c1 = concat(src1, gather(tar1, idx1))
// ---------------------------------------------------------------------------
__global__ void out_part1_kernel(const float* __restrict__ src1,
                                 const float* __restrict__ tar1,
                                 const int* __restrict__ idx1,
                                 float* __restrict__ out) {
  const int C = 1024, N = 4096;
  size_t gid = (size_t)blockIdx.x * blockDim.x + threadIdx.x;
  size_t total = (size_t)4 * 2 * C * N;
  if (gid >= total) return;
  int p  = (int)(gid % N);
  int ch = (int)((gid / N) % (2 * C));
  int b  = (int)(gid / ((size_t)N * 2 * C));
  float v;
  if (ch < C) v = src1[((size_t)b * C + ch) * N + p];
  else        v = tar1[((size_t)b * C + (ch - C)) * N + idx1[(size_t)b * N + p]];
  out[((size_t)b * 6144 + ch) * N + p] = v;
}

// ---------------------------------------------------------------------------
// Output channels [2048, 6144): bilinear 32->64 (align_corners=False) of
// concat(src2, gather(tar2, idx2)), gathered on the fly.
// ---------------------------------------------------------------------------
__global__ void out_part2_kernel(const float* __restrict__ src2,
                                 const float* __restrict__ tar2,
                                 const int* __restrict__ idx2,
                                 float* __restrict__ out) {
  const int C = 2048, Ns = 1024, Ws = 32, Wo = 64, Ho = 64;
  size_t gid = (size_t)blockIdx.x * blockDim.x + threadIdx.x;
  size_t total = (size_t)4 * 2 * C * Ho * Wo;
  if (gid >= total) return;
  int x  = (int)(gid % Wo);
  int y  = (int)((gid / Wo) % Ho);
  int ch = (int)((gid / ((size_t)Wo * Ho)) % (2 * C));
  int b  = (int)(gid / ((size_t)Wo * Ho * 2 * C));

  float syf = y * 0.5f - 0.25f;
  float sxf = x * 0.5f - 0.25f;
  int y0 = (int)floorf(syf);  float wy = syf - (float)y0;
  int x0 = (int)floorf(sxf);  float wx = sxf - (float)x0;
  int y1 = y0 + 1, x1 = x0 + 1;
  y0 = max(0, min(31, y0)); y1 = max(0, min(31, y1));
  x0 = max(0, min(31, x0)); x1 = max(0, min(31, x1));

  const int* id = idx2 + (size_t)b * Ns;
  auto fetch = [&](int yy, int xx) -> float {
    int q = yy * Ws + xx;
    if (ch < C) return src2[((size_t)b * C + ch) * Ns + q];
    return tar2[((size_t)b * C + (ch - C)) * Ns + id[q]];
  };
  float v = (1.f - wy) * ((1.f - wx) * fetch(y0, x0) + wx * fetch(y0, x1))
          +        wy  * ((1.f - wx) * fetch(y1, x0) + wx * fetch(y1, x1));
  out[((size_t)b * 6144 + 2048 + ch) * (Ho * Wo) + y * Wo + x] = v;
}

// ---------------------------------------------------------------------------
extern "C" void kernel_launch(void* const* d_in, const int* in_sizes, int n_in,
                              void* d_out, int out_size, void* d_ws, size_t ws_size,
                              hipStream_t stream) {
  const float* src1 = (const float*)d_in[0];  // [4,1024,64,64]
  const float* tar1 = (const float*)d_in[1];
  const float* src2 = (const float*)d_in[2];  // [4,2048,32,32]
  const float* tar2 = (const float*)d_in[3];
  float* out = (float*)d_out;                 // [4,6144,64,64]

  const int B = 4;
  const int C1 = 1024, N1 = 4096;
  const int C2 = 2048, N2 = 1024;

  // workspace: f16 transposed copies (~100 MB) + norms + indices
  _Float16* s1T = (_Float16*)d_ws;                   // [B][N1][C1]
  _Float16* t1T = s1T + (size_t)B * N1 * C1;
  _Float16* s2T = t1T + (size_t)B * N1 * C1;         // [B][N2][C2]
  _Float16* t2T = s2T + (size_t)B * N2 * C2;
  float* tn1 = (float*)(t2T + (size_t)B * N2 * C2);
  float* tn2 = tn1 + (size_t)B * N1;
  int* idx1  = (int*)(tn2 + (size_t)B * N2);
  int* idx2  = idx1 + (size_t)B * N1;

  // transpose + convert
  dim3 tg1(N1 / 32, C1 / 32, B);
  transpose_f16_kernel<<<tg1, 256, 0, stream>>>(src1, s1T, C1, N1);
  transpose_f16_kernel<<<tg1, 256, 0, stream>>>(tar1, t1T, C1, N1);
  dim3 tg2(N2 / 32, C2 / 32, B);
  transpose_f16_kernel<<<tg2, 256, 0, stream>>>(src2, s2T, C2, N2);
  transpose_f16_kernel<<<tg2, 256, 0, stream>>>(tar2, t2T, C2, N2);

  // target squared norms (f32 accuracy)
  tnorm_kernel<<<(B * N1 + 255) / 256, 256, 0, stream>>>(tar1, tn1, B, C1, N1);
  tnorm_kernel<<<(B * N2 + 255) / 256, 256, 0, stream>>>(tar2, tn2, B, C2, N2);

  // WMMA argmin
  dim3 g1(N1 / 128, B);
  nn_argmin_kernel<<<g1, 256, 0, stream>>>(s1T, t1T, tn1, idx1, C1, N1);
  dim3 g2(N2 / 128, B);
  nn_argmin_kernel<<<g2, 256, 0, stream>>>(s2T, t2T, tn2, idx2, C2, N2);

  // epilogues
  size_t tot1 = (size_t)B * 2 * C1 * N1;
  out_part1_kernel<<<(tot1 + 255) / 256, 256, 0, stream>>>(src1, tar1, idx1, out);
  size_t tot2 = (size_t)B * 2 * C2 * 64 * 64;
  out_part2_kernel<<<(tot2 + 255) / 256, 256, 0, stream>>>(src2, tar2, idx2, out);
}
